// TractBundle_48301202211096
// MI455X (gfx1250) — matrix operand order repaired and checked by reading, not compiled, not run on hardware
//
#include <hip/hip_runtime.h>

typedef __bf16 v16bf __attribute__((ext_vector_type(16)));
typedef __bf16 v8bf  __attribute__((ext_vector_type(8)));
typedef __bf16 v4bf  __attribute__((ext_vector_type(4)));
typedef float  v8f   __attribute__((ext_vector_type(8)));
typedef float  v16f  __attribute__((ext_vector_type(16)));
typedef float  v4f   __attribute__((ext_vector_type(4)));
typedef int    v4i   __attribute__((ext_vector_type(4)));
typedef unsigned short u16;

typedef __attribute__((address_space(1))) v4i* global_v4i_p;
typedef __attribute__((address_space(3))) v4i* lds_v4i_p;

#define BM 128
#define BN 256
#define BK 32
#define BKP 40   // padded LDS row stride (elements): 80B keeps 16B align, avoids bank conflicts

// ---- gfx1250 async global->LDS path (ASYNCcnt) ----
__device__ __forceinline__ void async_b128(const void* g, void* l) {
#if __has_builtin(__builtin_amdgcn_global_load_async_to_lds_b128)
    __builtin_amdgcn_global_load_async_to_lds_b128(
        (global_v4i_p)(uintptr_t)g,
        (lds_v4i_p)(unsigned)(uintptr_t)l,
        0, 0);
#else
    *(uint4*)l = *(const uint4*)g;   // sync fallback through VGPRs
#endif
}

__device__ __forceinline__ void wait_async0() {
#if __has_builtin(__builtin_amdgcn_s_wait_asynccnt)
    __builtin_amdgcn_s_wait_asynccnt(0);
#else
    asm volatile("s_wait_asynccnt 0x0" ::: "memory");
#endif
}

// Fold mask into weight once, store bf16 (hardware RNE convert) into workspace.
__global__ void tract_prep_wm(const float* __restrict__ w,
                              const float* __restrict__ m,
                              u16* __restrict__ wm, int n) {
    int i = (blockIdx.x * blockDim.x + threadIdx.x) * 4;
    if (i < n) {
        v4f wv = *(const v4f*)(w + i);
        v4f mv = *(const v4f*)(m + i);
        union { v4bf b; uint2 v; } r;
        r.b = __builtin_convertvector(wv * mv, v4bf);
        *(uint2*)(wm + i) = r.v;
    }
}

// C[m, colOff+n] = sum_k X[m,k] * Wm[n,k]   (Wm is bf16, pre-masked)
__global__ __launch_bounds__(256) void tract_gemm_bf16(
        const float* __restrict__ X, const u16* __restrict__ Wm,
        float* __restrict__ Out, int K, int ldOut, int colOff) {
    __shared__ u16 As[2][BM][BKP];   // 20 KB
    __shared__ u16 Bs[2][BN][BKP];   // 40 KB

    const int tid   = threadIdx.x;
    const int lane  = tid & 31;
    const int wave  = tid >> 5;
    const int waveM = wave & 1;    // 2 waves along M  -> 64 rows each
    const int waveN = wave >> 1;   // 4 waves along N  -> 64 cols each
    const int m0 = blockIdx.y * BM;
    const int n0 = blockIdx.x * BN;

    // cooperative A-tile mapping: each thread owns half a 32-wide row
    const int lr = tid >> 1;          // 0..127
    const int lc = (tid & 1) * 16;    // 0 or 16

    v8f acc[4][4];
    #pragma unroll
    for (int i = 0; i < 4; ++i)
        #pragma unroll
        for (int j = 0; j < 4; ++j)
            #pragma unroll
            for (int e = 0; e < 8; ++e) acc[i][j][e] = 0.0f;

    auto load_tiles = [&](int kt, int buf) {
        // A tile: 128x32 fp32 -> bf16 via v_cvt_pk_bf16_f32, then LDS
        const float* ap = X + (size_t)(m0 + lr) * K + (size_t)kt * BK + lc;
        union { v16f f; float4 q[4]; } a;
        a.q[0] = *(const float4*)(ap + 0);
        a.q[1] = *(const float4*)(ap + 4);
        a.q[2] = *(const float4*)(ap + 8);
        a.q[3] = *(const float4*)(ap + 12);
        union { v16bf b; uint4 v[2]; } r;
        r.b = __builtin_convertvector(a.f, v16bf);
        *(uint4*)&As[buf][lr][lc]     = r.v[0];
        *(uint4*)&As[buf][lr][lc + 8] = r.v[1];
        // B tile: 256x32 bf16, one full row per thread, DMA straight to LDS
        const u16* bp = Wm + (size_t)(n0 + tid) * K + (size_t)kt * BK;
        async_b128(bp,      &Bs[buf][tid][0]);
        async_b128(bp + 8,  &Bs[buf][tid][8]);
        async_b128(bp + 16, &Bs[buf][tid][16]);
        async_b128(bp + 24, &Bs[buf][tid][24]);
        // L2 prefetch one more k-tile ahead (global_prefetch_b8)
        __builtin_prefetch(ap + BK, 0, 0);
        __builtin_prefetch(bp + BK, 0, 0);
    };

    auto compute = [&](int buf) {
        const int mrow = lane & 15;
        const int ksel = (lane >> 4) & 1;
        v16bf af[4], bfr[4];
        #pragma unroll
        for (int i = 0; i < 4; ++i) {
            // A 16x32 bf16 layout: lanes 0-15 K in [0..7]∪[16..23]; lanes 16-31 K in [8..15]∪[24..31]
            const u16* p = &As[buf][waveM * 64 + i * 16 + mrow][ksel * 8];
            v8bf lo = *(const v8bf*)p;
            v8bf hi = *(const v8bf*)(p + 16);
            af[i] = __builtin_shufflevector(lo, hi, 0,1,2,3,4,5,6,7,8,9,10,11,12,13,14,15);
        }
        #pragma unroll
        for (int j = 0; j < 4; ++j) {
            // B 32x16 bf16 layout: lane n = lane%16; lanes 0-15 K in [0..7]∪[16..23]; lanes 16-31 K in [8..15]∪[24..31]
            const u16* p = &Bs[buf][waveN * 64 + j * 16 + mrow][ksel * 8];
            v8bf lo = *(const v8bf*)p;
            v8bf hi = *(const v8bf*)(p + 16);
            bfr[j] = __builtin_shufflevector(lo, hi, 0,1,2,3,4,5,6,7,8,9,10,11,12,13,14,15);
        }
        #pragma unroll
        for (int i = 0; i < 4; ++i)
            #pragma unroll
            for (int j = 0; j < 4; ++j)
                acc[i][j] = __builtin_amdgcn_wmma_f32_16x16x32_bf16(
                    false, af[i], false, bfr[j], (short)0, acc[i][j], false, false);
    };

    const int nk = K / BK;
    load_tiles(0, 0);
    wait_async0();
    __syncthreads();
    for (int kt = 0; kt < nk; ++kt) {
        const int buf = kt & 1;
        if (kt + 1 < nk) load_tiles(kt + 1, buf ^ 1);   // DMA overlaps compute below
        compute(buf);
        wait_async0();
        __syncthreads();
    }

    // C/D layout: VGPR r -> row r + 8*(lane/16), col lane%16
    #pragma unroll
    for (int i = 0; i < 4; ++i) {
        const int mBase = m0 + waveM * 64 + i * 16 + ((lane >> 4) << 3);
        #pragma unroll
        for (int j = 0; j < 4; ++j) {
            const int n = n0 + waveN * 64 + j * 16 + (lane & 15);
            float* op = Out + (size_t)mBase * ldOut + colOff + n;
            #pragma unroll
            for (int r = 0; r < 8; ++r)
                op[(size_t)r * ldOut] = acc[i][j][r];
        }
    }
}

extern "C" void kernel_launch(void* const* d_in, const int* in_sizes, int n_in,
                              void* d_out, int out_size, void* d_ws, size_t ws_size,
                              hipStream_t stream) {
    // setup_inputs() insertion order: x_a, w_a, m_a, x_b, w_b, m_b, x_c, w_c, m_c
    const float* x_a = (const float*)d_in[0];
    const float* w_a = (const float*)d_in[1];
    const float* m_a = (const float*)d_in[2];
    const float* x_b = (const float*)d_in[3];
    const float* w_b = (const float*)d_in[4];
    const float* m_b = (const float*)d_in[5];
    const float* x_c = (const float*)d_in[6];
    const float* w_c = (const float*)d_in[7];
    const float* m_c = (const float*)d_in[8];
    float* out = (float*)d_out;

    const int nA = 1024 * 4096, nB = 512 * 4096, nC = 512 * 2048;
    u16* wmA = (u16*)d_ws;
    u16* wmB = wmA + nA;
    u16* wmC = wmB + nB;

    tract_prep_wm<<<nA / 1024, 256, 0, stream>>>(w_a, m_a, wmA, nA);
    tract_prep_wm<<<nB / 1024, 256, 0, stream>>>(w_b, m_b, wmB, nB);
    tract_prep_wm<<<nC / 1024, 256, 0, stream>>>(w_c, m_c, wmC, nC);

    const int Mrows = 8 * 1024;       // B*S
    const int ldOut = 2048;           // 1024 + 512 + 512 concat
    tract_gemm_bf16<<<dim3(1024 / BN, Mrows / BM), 256, 0, stream>>>(
        x_a, wmA, out, 4096, ldOut, 0);
    tract_gemm_bf16<<<dim3(512 / BN, Mrows / BM), 256, 0, stream>>>(
        x_b, wmB, out, 4096, ldOut, 1024);
    tract_gemm_bf16<<<dim3(512 / BN, Mrows / BM), 256, 0, stream>>>(
        x_c, wmC, out, 2048, ldOut, 1536);
}
